// MultiHeadAttentioner_32942399160739
// MI455X (gfx1250) — matrix operand order repaired and checked by reading, not compiled, not run on hardware
//
#include <hip/hip_runtime.h>
#include <hip/hip_bf16.h>

// Shapes (compile-time, from the reference)
#define BB 2
#define SS 2048
#define EE 1024
#define HH 16
#define DH 64
#define MM (BB * SS)   // 4096 rows for all big GEMMs
#define LN_EPS 1e-5f

typedef __attribute__((ext_vector_type(16))) __bf16 v16bf;
typedef __attribute__((ext_vector_type(8)))  float  v8f;
typedef __attribute__((ext_vector_type(4)))  unsigned int u32x4;
typedef __attribute__((ext_vector_type(4)))  int i32x4;
typedef __attribute__((ext_vector_type(8)))  int i32x8;

union V16U { uint4 u[2]; v16bf v; };

#if __has_include(<hip/amd_detail/amd_gfx1250_TDM.h>)
#define TDM_6ARG 1
#else
#define TDM_6ARG 0
#endif

// ---------------------------------------------------------------------------
// WMMA helper. CDNA5 V_WMMA_F32_16X16X32_BF16, D = A(16x32) * B(32x16) + C.
// ---------------------------------------------------------------------------
__device__ __forceinline__ v8f wmma_bf16(v16bf a, v16bf b, v8f c) {
  return __builtin_amdgcn_wmma_f32_16x16x32_bf16(
      /*neg_a=*/false, a, /*neg_b=*/false, b,
      /*c_mod=*/(short)0, c, /*reuse_a=*/false, /*reuse_b=*/false);
}

// ---------------------------------------------------------------------------
// Tensor Data Mover: 2D bf16 tile Global -> LDS.  D# per CDNA5 ISA §8.3/8.4:
//   group0: count=1 | lds_addr | global_addr(57b) | type=2
//   group1: data_size=1(2B) | tensor_dim0/1 | tile_dim0/1 | dim0 stride
// groups 2/3 unused (2D tensor). All inputs must be wave-uniform (SGPRs).
// ---------------------------------------------------------------------------
__device__ __forceinline__ void tdm_load_2d_bf16(unsigned int lds_addr, const void* gaddr,
                                                 unsigned int tile_d0, unsigned int tile_d1,
                                                 unsigned int tensor_d0, unsigned int tensor_d1,
                                                 unsigned int stride0_elems) {
  unsigned long long ga = (unsigned long long)(uintptr_t)gaddr;
  u32x4 g0;
  g0[0] = 1u;                                             // count=1, user mode
  g0[1] = lds_addr;                                       // LDS byte address
  g0[2] = (unsigned int)(ga & 0xffffffffu);               // global_addr[31:0]
  g0[3] = (unsigned int)((ga >> 32) & 0x01ffffffu)        // global_addr[56:32]
          | (2u << 30);                                   // type=2 ("image")
  i32x8 g1;
  g1[0] = (int)(1u << 16);                                // data_size=1 (2B), mask=0
  g1[1] = (int)((tensor_d0 & 0xffffu) << 16);             // tensor_dim0[15:0]
  g1[2] = (int)((tensor_d0 >> 16) | ((tensor_d1 & 0xffffu) << 16));
  g1[3] = (int)((tensor_d1 >> 16) | ((tile_d0 & 0xffffu) << 16));
  g1[4] = (int)(tile_d1 & 0xffffu);                       // tile_dim1, tile_dim2=0
  g1[5] = (int)stride0_elems;                             // tensor_dim0_stride[31:0]
  g1[6] = 0;
  g1[7] = 0;
  i32x4 z4 = {0, 0, 0, 0};
#if TDM_6ARG
  i32x8 z8 = {0, 0, 0, 0, 0, 0, 0, 0};
  __builtin_amdgcn_tensor_load_to_lds(g0, g1, z4, z4, z8, 0);
#else
  __builtin_amdgcn_tensor_load_to_lds(g0, g1, z4, z4, 0);
#endif
}

// ---------------------------------------------------------------------------
// A-matrix tile (16x32, row-major source, ld in elements).
// ISA 16-bit A layout: lane (m = lane&15, half = lane>>4) holds
//   elems 0..7  -> K = k0 + 8*half + j ; elems 8..15 -> K = k0 + 16 + 8*half + j
// ---------------------------------------------------------------------------
__device__ __forceinline__ v16bf load_tile_A(const __bf16* base, int row0, int k0, int ld) {
  int lane = threadIdx.x & 31;
  int half = lane >> 4;
  int m    = lane & 15;
  const __bf16* p = base + (size_t)(row0 + m) * ld + k0 + 8 * half;
  V16U t;
  t.u[0] = *(const uint4*)(p);
  t.u[1] = *(const uint4*)(p + 16);
  return t.v;
}

// B-matrix tile (32x16) from a row-major [N,K] matrix (per-lane K contiguous).
// ISA 16-bit B layout: lane (n = lane&15, half) elems 0..7 -> K = k0 + 8*half + j ; elems 8..15 -> K = k0 + 16 + 8*half + j.
__device__ __forceinline__ v16bf load_tile_B(const __bf16* base, int n0, int k0, int ld) {
  int lane = threadIdx.x & 31;
  int half = lane >> 4;
  int n    = lane & 15;
  const __bf16* p = base + (size_t)(n0 + n) * ld + k0 + 8 * half;
  V16U t;
  t.u[0] = *(const uint4*)(p);
  t.u[1] = *(const uint4*)(p + 16);
  return t.v;
}

// ---------------------------------------------------------------------------
// f32 -> bf16 conversion
// ---------------------------------------------------------------------------
__global__ void k_f32_to_bf16(const float* __restrict__ src, __bf16* __restrict__ dst, int n) {
  int i = (blockIdx.x * blockDim.x + threadIdx.x) * 4;
  if (i + 3 < n) {
    float4 f = *(const float4*)(src + i);
    dst[i + 0] = (__bf16)f.x;
    dst[i + 1] = (__bf16)f.y;
    dst[i + 2] = (__bf16)f.z;
    dst[i + 3] = (__bf16)f.w;
  }
}

// ---------------------------------------------------------------------------
// GEMM: C[m,n] = sum_k A[m,k] * W[n,k] + bias[n]
// Block = 256 threads (8 waves), one 16-row slab of A staged to LDS via TDM;
// each wave owns one 16x16 N-tile. Grid (N/128, M/16).
// MODE 0: bf16 -> [B,H,S,Dh]; MODE 1: f32 -> [MM,EE]; MODE 2: bf16 -> [B,H,Dh,S]
// ---------------------------------------------------------------------------
template <int MODE>
__global__ void k_gemm(const __bf16* __restrict__ A, const __bf16* __restrict__ W,
                       const float* __restrict__ bias, void* __restrict__ outp) {
  __shared__ __align__(128) __bf16 sA[16 * EE];   // 32 KB slab

  int wave  = threadIdx.x >> 5;
  int ntile = blockIdx.x * 8 + wave;
  int row0  = blockIdx.y * 16;
  int n0    = ntile * 16;

  // Wave 0 DMAs the 16xEE bf16 A-slab into LDS; TENSORcnt is per-wave.
  if (wave == 0) {
    unsigned int lds_addr = (unsigned int)(uintptr_t)&sA[0];
    tdm_load_2d_bf16(lds_addr, A + (size_t)row0 * EE,
                     /*tile_d0=*/EE, /*tile_d1=*/16,
                     /*tensor_d0=*/EE, /*tensor_d1=*/16, /*stride0=*/EE);
    __builtin_amdgcn_s_wait_tensorcnt(0);
  }
  __syncthreads();

  int lane = threadIdx.x & 31;
  const __bf16* Wrow = W + (size_t)(n0 + (lane & 15)) * EE + 16 * (lane >> 4);

  v8f acc = {};
#pragma unroll 4
  for (int k0 = 0; k0 < EE; k0 += 32) {
    v16bf a = load_tile_A(sA, 0, k0, EE);   // ds_load_b128 from LDS slab
    v16bf b = load_tile_B(W, n0, k0, EE);   // global_load_b128
    if (k0 + 64 < EE) __builtin_prefetch((const void*)(Wrow + k0 + 64), 0, 1);
    acc = wmma_bf16(a, b, acc);
  }

  int half = lane >> 4;
  int n    = n0 + (lane & 15);
  float bn = bias[n];
#pragma unroll
  for (int r = 0; r < 8; ++r) {
    int m = row0 + r + 8 * half;
    float val = acc[r] + bn;
    if (MODE == 1) {
      ((float*)outp)[(size_t)m * EE + n] = val;
    } else {
      int b_ = m >> 11;
      int s  = m & (SS - 1);
      int h  = n >> 6;
      int d  = n & (DH - 1);
      size_t idx;
      if (MODE == 0) idx = (((size_t)(b_ * HH + h)) * SS + s) * DH + d;
      else           idx = (((size_t)(b_ * HH + h)) * DH + d) * SS + s;
      ((__bf16*)outp)[idx] = (__bf16)val;
    }
  }
}

// ---------------------------------------------------------------------------
// Flash attention (causal). One wave = one 16-row Q tile of one (b,h).
// __launch_bounds__(128, 1): wave32 allows up to 1024 VGPRs/wave — keep the
// Q tiles, 4 ctx accumulators, both score tiles and softmax state resident
// (round 2 showed scratch spill/reload of the Q tiles in the inner loop).
// Row sums of P come from a WMMA against a ones-matrix; only the row-max
// needs cross-lane shuffles.
// ---------------------------------------------------------------------------
__global__ void __launch_bounds__(128, 1)
k_attn(const __bf16* __restrict__ Q, const __bf16* __restrict__ K,
       const __bf16* __restrict__ Vt, __bf16* __restrict__ att) {
  __shared__ __align__(64) __bf16 pbuf[4][16 * 32];

  int wave = threadIdx.x >> 5;
  int lane = threadIdx.x & 31;
  int half = lane >> 4;
  int g     = blockIdx.x * 4 + wave;
  int bh    = g >> 7;
  int stile = g & 127;

  const __bf16* Qb = Q  + (size_t)bh * SS * DH;
  const __bf16* Kb = K  + (size_t)bh * SS * DH;
  const __bf16* Vb = Vt + (size_t)bh * DH * SS;

  v16bf aq0 = load_tile_A(Qb, stile * 16, 0,  DH);
  v16bf aq1 = load_tile_A(Qb, stile * 16, 32, DH);

  v16bf bones;
#pragma unroll
  for (int j = 0; j < 16; ++j) bones[j] = (__bf16)1.0f;

  v8f acc0 = {}, acc1 = {}, acc2 = {}, acc3 = {};
  float mrow[8], lrow[8];
#pragma unroll
  for (int r = 0; r < 8; ++r) { mrow[r] = -1e30f; lrow[r] = 0.0f; }

  const float scale = 0.125f;               // 1/sqrt(Dh)
  int nt = ((stile + 1) * 16 + 31) >> 5;    // causal T extent in 32-steps
  __bf16* pb = pbuf[wave];

  for (int tt = 0; tt < nt; ++tt) {
    int t0 = tt * 32;
    v8f s0 = {}, s1 = {};
    s0 = wmma_bf16(aq0, load_tile_B(Kb, t0,      0,  DH), s0);
    s0 = wmma_bf16(aq1, load_tile_B(Kb, t0,      32, DH), s0);
    s1 = wmma_bf16(aq0, load_tile_B(Kb, t0 + 16, 0,  DH), s1);
    s1 = wmma_bf16(aq1, load_tile_B(Kb, t0 + 16, 32, DH), s1);

    int col0 = t0 + (lane & 15);
#pragma unroll
    for (int r = 0; r < 8; ++r) {
      int rowg = stile * 16 + r + 8 * half;
      float v0 = (col0      <= rowg) ? s0[r] * scale : -1e30f;
      float v1 = (col0 + 16 <= rowg) ? s1[r] * scale : -1e30f;
      float mx = fmaxf(v0, v1);
      mx = fmaxf(mx, __shfl_xor(mx, 1));
      mx = fmaxf(mx, __shfl_xor(mx, 2));
      mx = fmaxf(mx, __shfl_xor(mx, 4));
      mx = fmaxf(mx, __shfl_xor(mx, 8));
      float mnew  = fmaxf(mrow[r], mx);
      float alpha = __expf(mrow[r] - mnew);
      float p0 = __expf(v0 - mnew);
      float p1 = __expf(v1 - mnew);
      mrow[r] = mnew;
      lrow[r] *= alpha;                      // fold alpha now: no alph[] array
      acc0[r] *= alpha; acc1[r] *= alpha; acc2[r] *= alpha; acc3[r] *= alpha;
      pb[(r + 8 * half) * 32 + (lane & 15)]      = (__bf16)p0;
      pb[(r + 8 * half) * 32 + (lane & 15) + 16] = (__bf16)p1;
    }

    v16bf ap = load_tile_A(pb, 0, 0, 32);
    // row sums: P(16x32) . ones(32x16) -> every column holds the row's sum
    v8f zs = {};
    v8f ls8 = wmma_bf16(ap, bones, zs);
#pragma unroll
    for (int r = 0; r < 8; ++r) lrow[r] += ls8[r];

    acc0 = wmma_bf16(ap, load_tile_B(Vb, 0,  t0, SS), acc0);
    acc1 = wmma_bf16(ap, load_tile_B(Vb, 16, t0, SS), acc1);
    acc2 = wmma_bf16(ap, load_tile_B(Vb, 32, t0, SS), acc2);
    acc3 = wmma_bf16(ap, load_tile_B(Vb, 48, t0, SS), acc3);
  }

  int b_ = bh >> 4, h = bh & 15;
  int n = lane & 15;
#pragma unroll
  for (int r = 0; r < 8; ++r) {
    int s = stile * 16 + r + 8 * half;
    float inv = 1.0f / lrow[r];
    size_t base = ((size_t)b_ * SS + s) * EE + h * DH;
    att[base + 0 * 16 + n] = (__bf16)(acc0[r] * inv);
    att[base + 1 * 16 + n] = (__bf16)(acc1[r] * inv);
    att[base + 2 * 16 + n] = (__bf16)(acc2[r] * inv);
    att[base + 3 * 16 + n] = (__bf16)(acc3[r] * inv);
  }
}

// ---------------------------------------------------------------------------
// residual + LayerNorm: out = LN(query + proj) * g + b. One block per row.
// ---------------------------------------------------------------------------
__global__ void k_ln(const float* __restrict__ query, const float* __restrict__ proj,
                     const float* __restrict__ gamma, const float* __restrict__ beta,
                     float* __restrict__ out) {
  int row = blockIdx.x;
  int tid = threadIdx.x;  // 256
  const float* qp = query + (size_t)row * EE;
  const float* pp = proj  + (size_t)row * EE;

  float x[4];
  float s = 0.f, s2 = 0.f;
#pragma unroll
  for (int j = 0; j < 4; ++j) {
    int i = tid + j * 256;
    x[j] = qp[i] + pp[i];
    s  += x[j];
    s2 += x[j] * x[j];
  }
  for (int m = 16; m >= 1; m >>= 1) { s += __shfl_xor(s, m); s2 += __shfl_xor(s2, m); }
  __shared__ float red[2][8];
  int w = tid >> 5;
  if ((tid & 31) == 0) { red[0][w] = s; red[1][w] = s2; }
  __syncthreads();
  if (w == 0) {
    float a  = (tid < 8) ? red[0][tid] : 0.f;
    float b2 = (tid < 8) ? red[1][tid] : 0.f;
    for (int m = 4; m >= 1; m >>= 1) { a += __shfl_xor(a, m); b2 += __shfl_xor(b2, m); }
    if (tid == 0) { red[0][0] = a; red[1][0] = b2; }
  }
  __syncthreads();
  float mean = red[0][0] * (1.0f / EE);
  float var  = red[1][0] * (1.0f / EE) - mean * mean;
  float rstd = rsqrtf(var + LN_EPS);
#pragma unroll
  for (int j = 0; j < 4; ++j) {
    int i = tid + j * 256;
    out[(size_t)row * EE + i] = (x[j] - mean) * rstd * gamma[i] + beta[i];
  }
}

// ---------------------------------------------------------------------------
// Launch
// ---------------------------------------------------------------------------
extern "C" void kernel_launch(void* const* d_in, const int* in_sizes, int n_in,
                              void* d_out, int out_size, void* d_ws, size_t ws_size,
                              hipStream_t stream) {
  const float* query = (const float*)d_in[0];
  const float* key_i = (const float*)d_in[1];
  const float* value = (const float*)d_in[2];
  // d_in[3] = mask: causal, computed analytically in-kernel
  const float* Wq_w = (const float*)d_in[4];
  const float* Wq_b = (const float*)d_in[5];
  const float* Wk_w = (const float*)d_in[6];
  const float* Wk_b = (const float*)d_in[7];
  const float* Wv_w = (const float*)d_in[8];
  const float* Wv_b = (const float*)d_in[9];
  const float* out_w = (const float*)d_in[10];
  const float* out_b = (const float*)d_in[11];
  const float* ln_g  = (const float*)d_in[12];
  const float* ln_b  = (const float*)d_in[13];
  float* out = (float*)d_out;

  char* ws = (char*)d_ws;
  const size_t MB = 1024 * 1024;
  __bf16* qb16 = (__bf16*)(ws + 0 * MB);    // [B,S,E] bf16      8MB
  __bf16* kb16 = (__bf16*)(ws + 8 * MB);
  __bf16* vb16 = (__bf16*)(ws + 16 * MB);
  __bf16* wq16 = (__bf16*)(ws + 24 * MB);   // [E,E] bf16        2MB
  __bf16* wk16 = (__bf16*)(ws + 26 * MB);
  __bf16* wv16 = (__bf16*)(ws + 28 * MB);
  __bf16* ow16 = (__bf16*)(ws + 30 * MB);
  __bf16* qpr  = (__bf16*)(ws + 32 * MB);   // [B,H,S,Dh] bf16   8MB
  __bf16* kpr  = (__bf16*)(ws + 40 * MB);   // [B,H,S,Dh] bf16   8MB
  __bf16* vprT = (__bf16*)(ws + 48 * MB);   // [B,H,Dh,S] bf16   8MB
  __bf16* att  = (__bf16*)(ws + 56 * MB);   // [B,S,E] bf16      8MB
  float*  proj = (float*)(ws + 64 * MB);    // [B,S,E] f32      16MB

  auto cvt = [&](const float* s, __bf16* d, int n) {
    k_f32_to_bf16<<<(n / 4 + 255) / 256, 256, 0, stream>>>(s, d, n);
  };
  cvt(query, qb16, MM * EE);
  cvt(key_i, kb16, MM * EE);
  cvt(value, vb16, MM * EE);
  cvt(Wq_w, wq16, EE * EE);
  cvt(Wk_w, wk16, EE * EE);
  cvt(Wv_w, wv16, EE * EE);
  cvt(out_w, ow16, EE * EE);

  dim3 gg(EE / 128, MM / 16);
  k_gemm<0><<<gg, 256, 0, stream>>>(qb16, wq16, Wq_b, qpr);
  k_gemm<0><<<gg, 256, 0, stream>>>(kb16, wk16, Wk_b, kpr);
  k_gemm<2><<<gg, 256, 0, stream>>>(vb16, wv16, Wv_b, vprT);

  k_attn<<<(BB * HH * (SS / 16)) / 4, 128, 0, stream>>>(qpr, kpr, vprT, att);

  k_gemm<1><<<gg, 256, 0, stream>>>(att, ow16, out_b, proj);
  k_ln<<<MM, 256, 0, stream>>>(query, proj, ln_g, ln_b, out);
}
